// MHSA_84146999263504
// MI455X (gfx1250) — matrix operand order repaired and checked by reading, not compiled, not run on hardware
//
#include <hip/hip_runtime.h>
#include <hip/hip_bf16.h>

typedef __bf16 bf16_t;
typedef __attribute__((ext_vector_type(16))) __bf16 v16bf;
typedef __attribute__((ext_vector_type(8)))  __bf16 v8bf;
typedef __attribute__((ext_vector_type(8)))  float  v8f;

#define BATCH 32
#define CCH   512
#define NHEAD 8
#define DH    64
#define WW    28
#define HH    28
#define NPIX  784     // 28*28
#define NXPAD 832     // Xt padded pixel count (13 * 64)
#define NVPAD 800     // V padded pixel count (25 * 32)

static __device__ __forceinline__ v16bf combine8(v8bf lo, v8bf hi) {
  v16bf r;
#pragma unroll
  for (int i = 0; i < 8; ++i) { r[i] = lo[i]; r[i + 8] = hi[i]; }
  return r;
}

static __device__ __forceinline__ v8f wmma_bf16(v16bf a, v16bf b, v8f c) {
  // D = A(16x32 bf16) * B(32x16 bf16) + C(16x16 f32)
  return __builtin_amdgcn_wmma_f32_16x16x32_bf16(false, a, false, b, (short)0, c,
                                                 false, false);
}

// ---------------------------------------------------------------------------
// Prep kernels: bf16 conversion + layout shuffles
// ---------------------------------------------------------------------------

__global__ void convert_w_kernel(const float* __restrict__ wq,
                                 const float* __restrict__ wk,
                                 const float* __restrict__ wv,
                                 bf16_t* __restrict__ Wbf) {
  int i = blockIdx.x * blockDim.x + threadIdx.x;
  const int per = CCH * CCH;
  if (i >= 3 * per) return;
  const float* src = (i < per) ? wq : (i < 2 * per ? wk : wv);
  Wbf[i] = (bf16_t)src[i % per];
}

// Pt[h][p][dd] = rel_h[h,dd,ih] + rel_w[h,dd,iw],  p = iw*HH + ih
__global__ void pos_kernel(const float* __restrict__ rel_h,
                           const float* __restrict__ rel_w,
                           bf16_t* __restrict__ Pt) {
  int i = blockIdx.x * blockDim.x + threadIdx.x;
  if (i >= NHEAD * NPIX * DH) return;
  int dd = i & 63;
  int p  = (i >> 6) % NPIX;
  int h  = i / (DH * NPIX);
  int iw = p / HH, ih = p % HH;
  float v = rel_h[(h * DH + dd) * HH + ih] + rel_w[(h * DH + dd) * WW + iw];
  Pt[i] = (bf16_t)v;
}

// Xt[b][n][c] = bf16(x[b][c][n]) via 32x32 LDS tile transpose
__global__ __launch_bounds__(256) void transpose_kernel(const float* __restrict__ x,
                                                        bf16_t* __restrict__ Xt) {
  __shared__ float tile[32][33];
  const int b = blockIdx.z, c0 = blockIdx.y * 32, n0 = blockIdx.x * 32;
  const int tx = threadIdx.x, ty = threadIdx.y;
#pragma unroll
  for (int yy = ty; yy < 32; yy += 8) {
    int n = n0 + tx;
    float v = (n < NPIX) ? x[((size_t)b * CCH + c0 + yy) * NPIX + n] : 0.f;
    tile[yy][tx] = v;
  }
  __syncthreads();
#pragma unroll
  for (int yy = ty; yy < 32; yy += 8) {
    int n = n0 + yy;  // <= 799 < NXPAD
    Xt[((size_t)b * NXPAD + n) * CCH + c0 + tx] = (bf16_t)tile[tx][yy];
  }
}

// ---------------------------------------------------------------------------
// QKV projection: Y[o,n] = sum_c W[o,c] * X[c,n] + bias[o]
// One wave -> 16(o) x 64(n) tile; WMMA K-loop over 512 channels.
// Outputs: Qt/Kt in [b][h][n][dd] (dd-contiguous), V in [b][h][dd][n].
// ---------------------------------------------------------------------------

__global__ __launch_bounds__(256) void qkv_proj_kernel(
    const bf16_t* __restrict__ Wbf, const bf16_t* __restrict__ Xt,
    const float* __restrict__ bq, const float* __restrict__ bk,
    const float* __restrict__ bv,
    bf16_t* __restrict__ Qt, bf16_t* __restrict__ Kt, bf16_t* __restrict__ Vd) {
  const int lane  = threadIdx.x & 31;
  const int wave  = threadIdx.x >> 5;               // 0..7
  const int nbase = blockIdx.x * 64;                // pixel tile base
  const int obase = (blockIdx.y * 8 + wave) * 16;   // out-channel tile base
  const int b     = blockIdx.z / 3;
  const int proj  = blockIdx.z % 3;                 // 0=q 1=k 2=v

  const int m   = lane & 15;          // A row / B column / D column
  const int kh  = lane >> 4;          // lane half
  const int ab0 = kh ? 16 : 0;        // A-fragment K offset
  const int bb0 = kh ? 16 : 0;        // B-fragment K offset

  const bf16_t* wrow  = Wbf + ((size_t)proj * CCH + obase + m) * CCH;
  const bf16_t* xbase = Xt + (size_t)b * NXPAD * CCH;

  v8f acc[4];
  const v8f vzero = {0.f, 0.f, 0.f, 0.f, 0.f, 0.f, 0.f, 0.f};
#pragma unroll
  for (int t = 0; t < 4; ++t) acc[t] = vzero;

  for (int kk = 0; kk < CCH; kk += 32) {
    const bf16_t* ap = wrow + kk + ab0;
    v16bf afrag = combine8(*(const v8bf*)ap, *(const v8bf*)(ap + 8));
#pragma unroll
    for (int t = 0; t < 4; ++t) {
      const bf16_t* bp = xbase + (size_t)(nbase + t * 16 + m) * CCH + kk + bb0;
      v16bf bfrag = combine8(*(const v8bf*)bp, *(const v8bf*)(bp + 8));
      acc[t] = wmma_bf16(afrag, bfrag, acc[t]);
    }
  }

  const float* bias = (proj == 0) ? bq : (proj == 1 ? bk : bv);
  const int o0  = obase + kh * 8;   // 8-run of out channels held by this lane
  const int h   = o0 >> 6;
  const int dd0 = o0 & 63;
#pragma unroll
  for (int t = 0; t < 4; ++t) {
    int n = nbase + t * 16 + m;
    if (n >= NPIX) continue;
    if (proj < 2) {
      bf16_t* dst = (proj == 0 ? Qt : Kt) +
                    (((size_t)(b * NHEAD + h) * NPIX + n) * DH + dd0);
      v8bf pack;
#pragma unroll
      for (int r = 0; r < 8; ++r) pack[r] = (bf16_t)(acc[t][r] + bias[o0 + r]);
      *(v8bf*)dst = pack;
    } else {
#pragma unroll
      for (int r = 0; r < 8; ++r) {
        Vd[((size_t)(b * NHEAD + h) * DH + dd0 + r) * NVPAD + n] =
            (bf16_t)(acc[t][r] + bias[o0 + r]);
      }
    }
  }
}

// ---------------------------------------------------------------------------
// Flash attention: scores = [Q;P]^T [K;Q]  (K-dim = 128), online softmax,
// O += softmax(S) * V^T.  One wave per 16-row i-tile; 7 waves per block.
// ---------------------------------------------------------------------------

__global__ __launch_bounds__(224) void attn_kernel(
    const bf16_t* __restrict__ Qt, const bf16_t* __restrict__ Kt,
    const bf16_t* __restrict__ Vd, const bf16_t* __restrict__ Pt,
    float* __restrict__ out) {
  __shared__ __align__(16) bf16_t pexp_lds[7][16 * 32];

  const int lane = threadIdx.x & 31;
  const int wave = threadIdx.x >> 5;   // 0..6
  int z = blockIdx.x;
  const int ig = z % 7; z /= 7;
  const int h  = z % NHEAD;
  const int b  = z / NHEAD;
  const int ibase = (ig * 7 + wave) * 16;

  const int m   = lane & 15;
  const int kh  = lane >> 4;
  const int ab0 = kh ? 16 : 0;
  const int bb0 = kh ? 16 : 0;

  const size_t bh  = (size_t)(b * NHEAD + h);
  const bf16_t* qtb = Qt + bh * NPIX * DH;
  const bf16_t* ktb = Kt + bh * NPIX * DH;
  const bf16_t* vdb = Vd + bh * DH * NVPAD;
  const bf16_t* ptb = Pt + (size_t)h * NPIX * DH;
  bf16_t* myLds = &pexp_lds[wave][0];

  // Loop-invariant A-fragments for the score GEMM: rows i, K = [Q(0..63); P(0..63)]
  const bf16_t* qrow = qtb + (size_t)(ibase + m) * DH;
  const bf16_t* prow = ptb + (size_t)(ibase + m) * DH;
  v16bf aQ0 = combine8(*(const v8bf*)(qrow + ab0),      *(const v8bf*)(qrow + ab0 + 8));
  v16bf aQ1 = combine8(*(const v8bf*)(qrow + 32 + ab0), *(const v8bf*)(qrow + 32 + ab0 + 8));
  v16bf aP0 = combine8(*(const v8bf*)(prow + ab0),      *(const v8bf*)(prow + ab0 + 8));
  v16bf aP1 = combine8(*(const v8bf*)(prow + 32 + ab0), *(const v8bf*)(prow + 32 + ab0 + 8));

  const v8f vzero = {0.f, 0.f, 0.f, 0.f, 0.f, 0.f, 0.f, 0.f};
  float mrun[8], lrun[8];
#pragma unroll
  for (int r = 0; r < 8; ++r) { mrun[r] = -3.0e38f; lrun[r] = 0.f; }
  v8f oacc[4];
#pragma unroll
  for (int t = 0; t < 4; ++t) oacc[t] = vzero;

  auto score_tile = [&](int jcol0) -> v8f {
    const int j = jcol0 + m;
    const bf16_t* krow  = ktb + (size_t)j * DH;
    const bf16_t* qjrow = qtb + (size_t)j * DH;
    v8f t = vzero;
    t = wmma_bf16(aQ0, combine8(*(const v8bf*)(krow + bb0),
                                *(const v8bf*)(krow + bb0 + 8)), t);
    t = wmma_bf16(aQ1, combine8(*(const v8bf*)(krow + 32 + bb0),
                                *(const v8bf*)(krow + 32 + bb0 + 8)), t);
    t = wmma_bf16(aP0, combine8(*(const v8bf*)(qjrow + bb0),
                                *(const v8bf*)(qjrow + bb0 + 8)), t);
    t = wmma_bf16(aP1, combine8(*(const v8bf*)(qjrow + 32 + bb0),
                                *(const v8bf*)(qjrow + 32 + bb0 + 8)), t);
    return t;
  };

  const int nblocks = (NPIX + 31) / 32;  // 25 (last block has 16 valid j)
  for (int jb = 0; jb < nblocks; ++jb) {
    const int jbase = jb * 32;
    const bool tail = (jbase + 32 > NPIX);   // uniform across the block

    v8f s0 = score_tile(jbase);
    v8f s1;
    if (!tail) {
      s1 = score_tile(jbase + 16);
    } else {
#pragma unroll
      for (int r = 0; r < 8; ++r) s1[r] = -1.0e30f;   // exp -> 0
    }

    // Online softmax; rows m = r + 8*kh, columns live across a 16-lane half.
#pragma unroll
    for (int r = 0; r < 8; ++r) {
      float rm = fmaxf(s0[r], s1[r]);
      rm = fmaxf(rm, __shfl_xor(rm, 1, 32));
      rm = fmaxf(rm, __shfl_xor(rm, 2, 32));
      rm = fmaxf(rm, __shfl_xor(rm, 4, 32));
      rm = fmaxf(rm, __shfl_xor(rm, 8, 32));
      float nm    = fmaxf(mrun[r], rm);
      float scale = __expf(mrun[r] - nm);
      float p0    = __expf(s0[r] - nm);
      float p1    = __expf(s1[r] - nm);
      float ps = p0 + p1;
      ps += __shfl_xor(ps, 1, 32);
      ps += __shfl_xor(ps, 2, 32);
      ps += __shfl_xor(ps, 4, 32);
      ps += __shfl_xor(ps, 8, 32);
      lrun[r] = lrun[r] * scale + ps;
      mrun[r] = nm;
#pragma unroll
      for (int dt = 0; dt < 4; ++dt) oacc[dt][r] *= scale;
      const int row = r + 8 * kh;
      myLds[row * 32 + m]      = (bf16_t)p0;
      myLds[row * 32 + 16 + m] = (bf16_t)p1;
    }
    __syncthreads();

    // Re-fragment Pexp from LDS as the A operand (16 x 32 over local j)
    v16bf pe = combine8(*(const v8bf*)(myLds + m * 32 + ab0),
                        *(const v8bf*)(myLds + m * 32 + 8 + ab0));
#pragma unroll
    for (int dt = 0; dt < 4; ++dt) {
      const bf16_t* vrow = vdb + (size_t)(dt * 16 + m) * NVPAD + jbase + bb0;
      v16bf vf = combine8(*(const v8bf*)vrow, *(const v8bf*)(vrow + 8));
      oacc[dt] = wmma_bf16(pe, vf, oacc[dt]);
    }
    __syncthreads();
  }

  // Epilogue: out[b, h*64+d, i] = O[i,d] / l[i]; rows contiguous in i per lane.
  float inv[8];
#pragma unroll
  for (int r = 0; r < 8; ++r) inv[r] = 1.0f / lrun[r];
  const int ibw = ibase + 8 * kh;
#pragma unroll
  for (int dt = 0; dt < 4; ++dt) {
    float* dst = out + ((size_t)(b * CCH + h * DH + dt * 16 + m)) * NPIX + ibw;
    float4 f0 = {oacc[dt][0] * inv[0], oacc[dt][1] * inv[1],
                 oacc[dt][2] * inv[2], oacc[dt][3] * inv[3]};
    float4 f1 = {oacc[dt][4] * inv[4], oacc[dt][5] * inv[5],
                 oacc[dt][6] * inv[6], oacc[dt][7] * inv[7]};
    *(float4*)(dst)     = f0;
    *(float4*)(dst + 4) = f1;
  }
}

// ---------------------------------------------------------------------------

extern "C" void kernel_launch(void* const* d_in, const int* in_sizes, int n_in,
                              void* d_out, int out_size, void* d_ws, size_t ws_size,
                              hipStream_t stream) {
  const float* x     = (const float*)d_in[0];
  const float* wq    = (const float*)d_in[1];
  const float* bq    = (const float*)d_in[2];
  const float* wk    = (const float*)d_in[3];
  const float* bk    = (const float*)d_in[4];
  const float* wv    = (const float*)d_in[5];
  const float* bv    = (const float*)d_in[6];
  const float* rel_h = (const float*)d_in[7];
  const float* rel_w = (const float*)d_in[8];
  float* out = (float*)d_out;
  (void)in_sizes; (void)n_in; (void)out_size; (void)ws_size;

  char* ws = (char*)d_ws;
  size_t off = 0;
  auto carve = [&](size_t bytes) -> char* {
    char* p = ws + off;
    off = (off + bytes + 255) & ~(size_t)255;
    return p;
  };
  bf16_t* Xt  = (bf16_t*)carve((size_t)BATCH * NXPAD * CCH * 2);          // 27.3 MB
  bf16_t* Wbf = (bf16_t*)carve((size_t)3 * CCH * CCH * 2);                //  1.6 MB
  bf16_t* Qt  = (bf16_t*)carve((size_t)BATCH * NHEAD * NPIX * DH * 2);    // 25.7 MB
  bf16_t* Kt  = (bf16_t*)carve((size_t)BATCH * NHEAD * NPIX * DH * 2);    // 25.7 MB
  bf16_t* Vd  = (bf16_t*)carve((size_t)BATCH * NHEAD * DH * NVPAD * 2);   // 26.2 MB
  bf16_t* Pt  = (bf16_t*)carve((size_t)NHEAD * NPIX * DH * 2);            //  0.8 MB

  convert_w_kernel<<<(3 * CCH * CCH + 255) / 256, 256, 0, stream>>>(wq, wk, wv, Wbf);
  pos_kernel<<<(NHEAD * NPIX * DH + 255) / 256, 256, 0, stream>>>(rel_h, rel_w, Pt);
  transpose_kernel<<<dim3(25, 16, BATCH), dim3(32, 8), 0, stream>>>(x, Xt);
  qkv_proj_kernel<<<dim3(13, 4, BATCH * 3), 256, 0, stream>>>(Wbf, Xt, bq, bk, bv,
                                                              Qt, Kt, Vd);
  attn_kernel<<<BATCH * NHEAD * 7, 224, 0, stream>>>(Qt, Kt, Vd, Pt, out);
}
